// LightGlueONNX_32006096290376
// MI455X (gfx1250) — compile-verified, matrix operand order checked by reading, not run on hardware
//
#include <hip/hip_runtime.h>
#include <stdint.h>

// ---------------------------------------------------------------------------
// CDNA5 (gfx1250) fused attention, bf16 WMMA (16x16x32), f32 accumulation,
// async global->LDS staging of K / V^T tiles shared by 4 waves per block.
// All inter-kernel tensors bf16; weights/descs pre-converted once.
// ---------------------------------------------------------------------------

typedef __attribute__((ext_vector_type(16))) __bf16        v16bf;
typedef __attribute__((ext_vector_type(8)))  float         v8f;
typedef __attribute__((ext_vector_type(8)))  unsigned int  v8u;

union FragB { v16bf bf; v8u u; };

#define DIM  256
#define NSEQ 4096
#define KSCL 0.09016844005555521f   /* (1/sqrt(256)) * log2(e) */

#define KSTRIDE 264                  /* bf16 elems per K-tile row (512B + 16B pad) */
#define VSTRIDE 40                   /* bf16 elems per V^T-tile row (64B + 16B pad) */
#define KBUFB   (32 * KSTRIDE * 2)   /* bytes per K buffer  = 16896 */
#define VBUFB   (256 * VSTRIDE * 2)  /* bytes per V buffer  = 20480 */

static __device__ __forceinline__ unsigned short bf16_rn(float x) {
  unsigned u = __float_as_uint(x);
  u += 0x7FFFu + ((u >> 16) & 1u);           // round-to-nearest-even
  return (unsigned short)(u >> 16);
}
static __device__ __forceinline__ unsigned pack2(float a, float b) {
  return (unsigned)bf16_rn(a) | ((unsigned)bf16_rn(b) << 16);
}

static __device__ __forceinline__ v8f wmma_bf16(const FragB& a, const FragB& b, v8f c) {
  return __builtin_amdgcn_wmma_f32_16x16x32_bf16(
      false, a.bf, false, b.bf, (short)0, c, false, false);
}

// 16-bit A/B fragment (16x32 / 32x16): per-lane = one row (A) / one column (B).
// lanes 0-15: u[0..3] <- K 0..7,  u[4..7] <- K 16..23
// lanes 16-31: u[0..3] <- K 8..15, u[4..7] <- K 24..31
static __device__ __forceinline__ FragB load_frag_bf16(const unsigned short* p, int half) {
  FragB f;
  uint4 x = *(const uint4*)(p + 8 * half);
  uint4 y = *(const uint4*)(p + 16 + 8 * half);
  f.u[0] = x.x; f.u[1] = x.y; f.u[2] = x.z; f.u[3] = x.w;
  f.u[4] = y.x; f.u[5] = y.y; f.u[6] = y.z; f.u[7] = y.w;
  return f;
}

// ---------------------------------------------------------------------------
// One-shot f32 -> bf16 conversion (descs + weights), 8 elems/thread.
// ---------------------------------------------------------------------------
__global__ void __launch_bounds__(256)
cvt_bf16_kernel(const float* __restrict__ in, unsigned short* __restrict__ out, int n8) {
  int i = blockIdx.x * 256 + threadIdx.x;
  if (i >= n8) return;
  const float4* p = (const float4*)(in + (size_t)i * 8);
  float4 a = p[0], b = p[1];
  uint4 v;
  v.x = pack2(a.x, a.y); v.y = pack2(a.z, a.w);
  v.z = pack2(b.x, b.y); v.w = pack2(b.z, b.w);
  *(uint4*)(out + (size_t)i * 8) = v;
}

// ---------------------------------------------------------------------------
// Projection: out[m,n] = sum_k X[m,k] * W[n,k] + b[n]   (torch Linear, NT form)
// X, W already bf16.  MODE 0: bf16 row-major (Q,K) | MODE 1: bf16 transposed
// [B,256,4096] (V^T) | MODE 2: f32 row-major (final Wo projection -> d_out)
// ---------------------------------------------------------------------------
template <int MODE>
__global__ void __launch_bounds__(128)
proj_kernel(const unsigned short* __restrict__ X, const unsigned short* __restrict__ W,
            const float* __restrict__ bias,
            unsigned short* __restrict__ outb, float* __restrict__ outf) {
  const int lane = threadIdx.x & 31;
  const int half = lane >> 4;
  const int mrow = lane & 15;
  const int wave = blockIdx.x * 4 + (threadIdx.x >> 5);
  const int tokBase = wave << 4;

  FragB xa[8];
  const unsigned short* xrow = X + (size_t)(tokBase + mrow) * DIM;
#pragma unroll
  for (int c = 0; c < 8; c++) xa[c] = load_frag_bf16(xrow + 32 * c, half);

  const unsigned short* wrow = W + (size_t)mrow * DIM;
#pragma unroll 1
  for (int t = 0; t < 16; t++) {
    const float bv = bias[16 * t + mrow];
    v8f acc;
#pragma unroll
    for (int r = 0; r < 8; r++) acc[r] = bv;

    const unsigned short* wp = wrow + (size_t)(16 * t) * DIM;
#pragma unroll
    for (int c = 0; c < 8; c++) {
      FragB wb = load_frag_bf16(wp + 32 * c, half);
      acc = wmma_bf16(xa[c], wb, acc);
    }

    if (MODE == 0) {
#pragma unroll
      for (int r = 0; r < 8; r++) {
        int tok = tokBase + r + 8 * half;
        outb[(size_t)tok * DIM + 16 * t + mrow] = bf16_rn(acc[r]);
      }
    } else if (MODE == 1) {
      int batch = tokBase >> 12;
      int tin   = (tokBase & (NSEQ - 1)) + 8 * half;
      int n     = 16 * t + mrow;
      unsigned short* dst =
          outb + (size_t)batch * DIM * NSEQ + (size_t)n * NSEQ + tin;
      uint4 v;
      v.x = pack2(acc[0], acc[1]); v.y = pack2(acc[2], acc[3]);
      v.z = pack2(acc[4], acc[5]); v.w = pack2(acc[6], acc[7]);
      *(uint4*)dst = v;
    } else {
#pragma unroll
      for (int r = 0; r < 8; r++) {
        int tok = tokBase + r + 8 * half;
        outf[(size_t)tok * DIM + 16 * t + mrow] = acc[r];
      }
    }
  }
}

// ---------------------------------------------------------------------------
// Fused flash attention with async-staged K / V^T tiles (double buffered).
// One wave = one 16-query tile; 4 waves/block share key tiles via LDS.
// ---------------------------------------------------------------------------
__global__ void __launch_bounds__(128)
attn_kernel(const unsigned short* __restrict__ Qb,
            const unsigned short* __restrict__ Kb,
            const unsigned short* __restrict__ Vt,
            unsigned short* __restrict__ Ob) {
  __shared__ __align__(16) unsigned short Ktile[2][32 * KSTRIDE];   // 2 x 16896 B
  __shared__ __align__(16) unsigned short Vtile[2][256 * VSTRIDE];  // 2 x 20480 B
  __shared__ __align__(16) unsigned short PldsAll[4][16 * 34];      // 4352 B

  const int lane = threadIdx.x & 31;
  const int wib  = threadIdx.x >> 5;
  const int half = lane >> 4;
  const int mrow = lane & 15;
  unsigned short* Plds = PldsAll[wib];

  const int qt    = blockIdx.x * 4 + wib;     // 0..1023 query tiles
  const int batch = qt >> 8;
  const int qBase = qt << 4;

  const unsigned short* qrow = Qb + (size_t)(qBase + mrow) * DIM;
  const unsigned long long gK =
      (unsigned long long)(Kb + (size_t)batch * NSEQ * DIM);
  const unsigned long long gV =
      (unsigned long long)(Vt + (size_t)batch * DIM * NSEQ);

  // Q tile lives in registers for the whole kernel (8 x 8 VGPRs).
  FragB qa[8];
#pragma unroll
  for (int c = 0; c < 8; c++) qa[c] = load_frag_bf16(qrow + 32 * c, half);

  // Per-lane 16B-chunk coordinates for the cooperative async copies.
  // K tile: 32 rows x 512B  = 1024 chunks; V tile: 256 rows x 64B = 1024 chunks.
  unsigned            kDst[8], vDst[8];
  unsigned long long  kSrc[8], vSrc[8];
#pragma unroll
  for (int j = 0; j < 8; j++) {
    int c = wib * 256 + j * 32 + lane;
    int kr = c >> 5, kc = c & 31;
    kDst[j] = (unsigned)(kr * (KSTRIDE * 2) + kc * 16);
    kSrc[j] = (unsigned long long)kr * (DIM * 2) + (unsigned long long)kc * 16;
    int vr = c >> 2, vc = c & 3;
    vDst[j] = (unsigned)(vr * (VSTRIDE * 2) + vc * 16);
    vSrc[j] = (unsigned long long)vr * (NSEQ * 2) + (unsigned long long)vc * 16;
  }
  const unsigned ldsK0 = (unsigned)(uintptr_t)&Ktile[0][0];
  const unsigned ldsV0 = (unsigned)(uintptr_t)&Vtile[0][0];

  auto stage = [&](int buf, int kb) {
    unsigned long long kbase = gK + (unsigned long long)kb * (DIM * 2);
    unsigned long long vbase = gV + (unsigned long long)kb * 2;
#pragma unroll
    for (int j = 0; j < 8; j++) {
      unsigned d = ldsK0 + (unsigned)buf * KBUFB + kDst[j];
      unsigned long long s = kbase + kSrc[j];
      asm volatile("global_load_async_to_lds_b128 %0, %1, off"
                   :: "v"(d), "v"(s) : "memory");
    }
#pragma unroll
    for (int j = 0; j < 8; j++) {
      unsigned d = ldsV0 + (unsigned)buf * VBUFB + vDst[j];
      unsigned long long s = vbase + vSrc[j];
      asm volatile("global_load_async_to_lds_b128 %0, %1, off"
                   :: "v"(d), "v"(s) : "memory");
    }
  };

  float mr[8], lr[8];
  v8f O[16];
#pragma unroll
  for (int r = 0; r < 8; r++) { mr[r] = -1e30f; lr[r] = 0.f; }
#pragma unroll
  for (int t = 0; t < 16; t++) {
#pragma unroll
    for (int r = 0; r < 8; r++) O[t][r] = 0.f;
  }

  stage(0, 0);                                // prologue prefetch
  int buf = 0;

#pragma unroll 1
  for (int kb = 0; kb < NSEQ; kb += 32) {
    if (kb + 32 < NSEQ) {
      stage(buf ^ 1, kb + 32);                // prefetch next tile
      asm volatile("s_wait_asynccnt 0x10" ::: "memory");  // older 16 done
    } else {
      asm volatile("s_wait_asynccnt 0x0" ::: "memory");
    }
    __syncthreads();                          // publish staged tile to all waves

    const unsigned short* KtB = &Ktile[buf][0];
    const unsigned short* VtL = &Vtile[buf][0];

    v8f S0, S1;
#pragma unroll
    for (int r = 0; r < 8; r++) { S0[r] = 0.f; S1[r] = 0.f; }

    // S = Q * K^T over D=256 in eight 32-wide chunks (K from LDS, Q in regs).
#pragma unroll
    for (int c = 0; c < 8; c++) {
      FragB k0 = load_frag_bf16(KtB + mrow * KSTRIDE + 32 * c, half);
      FragB k1 = load_frag_bf16(KtB + (16 + mrow) * KSTRIDE + 32 * c, half);
      S0 = wmma_bf16(qa[c], k0, S0);
      S1 = wmma_bf16(qa[c], k1, S1);
    }

    // Online softmax (base-2 domain, scale folded into KSCL).
    float fs[8];
#pragma unroll
    for (int r = 0; r < 8; r++) {
      float s0 = S0[r] * KSCL, s1 = S1[r] * KSCL;
      float cm = fmaxf(s0, s1);
      cm = fmaxf(cm, __shfl_xor(cm, 1));
      cm = fmaxf(cm, __shfl_xor(cm, 2));
      cm = fmaxf(cm, __shfl_xor(cm, 4));
      cm = fmaxf(cm, __shfl_xor(cm, 8));
      float mn = fmaxf(mr[r], cm);
      float p0 = exp2f(s0 - mn);
      float p1 = exp2f(s1 - mn);
      float f  = exp2f(mr[r] - mn);
      float sm = p0 + p1;
      sm += __shfl_xor(sm, 1);
      sm += __shfl_xor(sm, 2);
      sm += __shfl_xor(sm, 4);
      sm += __shfl_xor(sm, 8);
      lr[r] = lr[r] * f + sm;
      mr[r] = mn;
      fs[r] = f;
      int row = r + 8 * half;
      Plds[row * 34 + mrow]      = bf16_rn(p0);
      Plds[row * 34 + 16 + mrow] = bf16_rn(p1);
    }
#pragma unroll
    for (int t = 0; t < 16; t++) {
#pragma unroll
      for (int r = 0; r < 8; r++) O[t][r] *= fs[r];
    }

    // Same-wave LDS ops are in-order; fence the compiler only.
    __builtin_amdgcn_wave_barrier();
    asm volatile("" ::: "memory");

    FragB pa;                                 // P as bf16 A-fragment
#pragma unroll
    for (int j = 0; j < 4; j++) {
      pa.u[j]     = *(const unsigned*)(Plds + mrow * 34 + 8 * half + 2 * j);
      pa.u[4 + j] = *(const unsigned*)(Plds + mrow * 34 + 16 + 8 * half + 2 * j);
    }

    // O += P * V : B-fragment lane = row of staged V^T tile.
#pragma unroll
    for (int t = 0; t < 16; t++) {
      FragB vb = load_frag_bf16(VtL + (16 * t + mrow) * VSTRIDE, half);
      O[t] = wmma_bf16(pa, vb, O[t]);
    }

    __syncthreads();                          // all reads of buf done before refill
    buf ^= 1;
  }

  // Normalize and store bf16 attention output (consumed by Wo projection).
#pragma unroll
  for (int r = 0; r < 8; r++) {
    float inv = 1.f / lr[r];
    int row = qBase + r + 8 * half;
    unsigned short* op = Ob + (size_t)row * DIM + mrow;
#pragma unroll
    for (int t = 0; t < 16; t++) op[16 * t] = bf16_rn(O[t][r] * inv);
  }
}

// ---------------------------------------------------------------------------
extern "C" void kernel_launch(void* const* d_in, const int* in_sizes, int n_in,
                              void* d_out, int out_size, void* d_ws, size_t ws_size,
                              hipStream_t stream) {
  const float* desc0 = (const float*)d_in[0];
  const float* desc1 = (const float*)d_in[1];
  const float* Wq = (const float*)d_in[2]; const float* bq = (const float*)d_in[3];
  const float* Wk = (const float*)d_in[4]; const float* bk = (const float*)d_in[5];
  const float* Wv = (const float*)d_in[6]; const float* bv = (const float*)d_in[7];
  const float* Wo = (const float*)d_in[8]; const float* bo = (const float*)d_in[9];
  float* out = (float*)d_out;

  const int    tokens = 4 * NSEQ;              // 16384
  const size_t TD     = (size_t)tokens * DIM;  // 4,194,304 elems
  unsigned short* X0b = (unsigned short*)d_ws;
  unsigned short* X1b = X0b + TD;
  unsigned short* Qb  = X1b + TD;
  unsigned short* Kb  = Qb  + TD;
  unsigned short* Vt  = Kb  + TD;
  unsigned short* Ab  = Vt  + TD;
  unsigned short* Wqb = Ab  + TD;
  unsigned short* Wkb = Wqb + DIM * DIM;
  unsigned short* Wvb = Wkb + DIM * DIM;
  unsigned short* Wob = Wvb + DIM * DIM;

  // One-shot bf16 conversion of descriptors and weights.
  {
    int n8d = (int)(TD / 8);                   // 524288
    int n8w = DIM * DIM / 8;                   // 8192
    dim3 cb(256);
    cvt_bf16_kernel<<<dim3((n8d + 255) / 256), cb, 0, stream>>>(desc0, X0b, n8d);
    cvt_bf16_kernel<<<dim3((n8d + 255) / 256), cb, 0, stream>>>(desc1, X1b, n8d);
    cvt_bf16_kernel<<<dim3((n8w + 255) / 256), cb, 0, stream>>>(Wq, Wqb, n8w);
    cvt_bf16_kernel<<<dim3((n8w + 255) / 256), cb, 0, stream>>>(Wk, Wkb, n8w);
    cvt_bf16_kernel<<<dim3((n8w + 255) / 256), cb, 0, stream>>>(Wv, Wvb, n8w);
    cvt_bf16_kernel<<<dim3((n8w + 255) / 256), cb, 0, stream>>>(Wo, Wob, n8w);
  }

  dim3 blk(128);
  dim3 grd(tokens / 16 / 4);                   // 256 blocks, 1024 waves

  proj_kernel<0><<<grd, blk, 0, stream>>>(X0b, Wqb, bq, Qb, nullptr);
  proj_kernel<0><<<grd, blk, 0, stream>>>(X1b, Wkb, bk, Kb, nullptr);
  proj_kernel<1><<<grd, blk, 0, stream>>>(X1b, Wvb, bv, Vt, nullptr);
  attn_kernel<<<grd, blk, 0, stream>>>(Qb, Kb, Vt, Ab);
  proj_kernel<2><<<grd, blk, 0, stream>>>(Ab, Wob, bo, nullptr, out);
}
